// GCLAU_83476984365520
// MI455X (gfx1250) — compile-verified, matrix operand order checked by reading, not run on hardware
//
#include <hip/hip_runtime.h>
#include <hip/hip_bf16.h>
#include <hip/hip_fp16.h>
#include <math.h>

#define NUM_USERS 200000
#define NUM_ITEMS 100000
#define N_NODES   (NUM_USERS + NUM_ITEMS)
#define EMBED     64
#define INNER     256
#define LAYERS    3
#define EPS_NOISE 0.1f
#define BATCH     2048

typedef __attribute__((ext_vector_type(16))) _Float16 v16h;
typedef __attribute__((ext_vector_type(8)))  _Float16 v8h;
typedef __attribute__((ext_vector_type(8)))  float    v8f;

// ---------------------------------------------------------------- utility
__global__ void k_zero(float* __restrict__ p, long n) {
    long i = (long)blockIdx.x * blockDim.x + threadIdx.x;
    if (i < n) p[i] = 0.f;
}

__global__ void k_concat(const float* __restrict__ ue, const float* __restrict__ ie,
                         float* __restrict__ e) {
    long i = (long)blockIdx.x * blockDim.x + threadIdx.x;
    const long nu = (long)NUM_USERS * EMBED;
    const long nn = (long)N_NODES * EMBED;
    if (i < nu)       e[i] = ue[i];
    else if (i < nn)  e[i] = ie[i - nu];
}

__global__ void k_scale(const float* __restrict__ in, float* __restrict__ out,
                        float s, long n) {
    long i = (long)blockIdx.x * blockDim.x + threadIdx.x;
    if (i < n) out[i] = in[i] * s;
}

__global__ void k_f32_to_f16(const float* __restrict__ in, _Float16* __restrict__ out, long n) {
    long i = (long)blockIdx.x * blockDim.x + threadIdx.x;
    if (i < n) out[i] = (_Float16)in[i];
}

// ---------------------------------------------------------------- SpMM (edge-parallel)
// one wave per edge; lane handles 2 of the 64 embedding components
__global__ void k_spmm(const int* __restrict__ row, const int* __restrict__ col,
                       const float* __restrict__ val, const float* __restrict__ e,
                       float* __restrict__ out, int nE) {
    long gt  = (long)blockIdx.x * blockDim.x + threadIdx.x;
    int edge = (int)(gt >> 5);
    if (edge >= nE) return;
    int lane = (int)(gt & 31);
    int r = row[edge], c = col[edge];
    float v = val[edge];
    const float2 s = *(const float2*)(e + (long)c * EMBED + lane * 2);
    float* dst = out + (long)r * EMBED + lane * 2;
    atomicAdd(dst,     v * s.x);
    atomicAdd(dst + 1, v * s.y);
}

__global__ void k_noise(float* __restrict__ e, const float* __restrict__ nz, long n) {
    long i = (long)blockIdx.x * blockDim.x + threadIdx.x;
    if (i >= n) return;
    float x = e[i];
    float sg = (x > 0.f) ? 1.f : ((x < 0.f) ? -1.f : 0.f);
    e[i] = x + sg * nz[i] * EPS_NOISE;
}

// acc[b][d] += e[(base+idx[b])*64 + d]
__global__ void k_gather_add(const float* __restrict__ e, const int* __restrict__ idx,
                             int base, float* __restrict__ acc) {
    int i = blockIdx.x * blockDim.x + threadIdx.x;
    if (i >= BATCH * EMBED) return;
    int b = i >> 6, d = i & 63;
    acc[i] += e[(long)(base + idx[b]) * EMBED + d];
}

// ---------------------------------------------------------------- row-wise ops (D = 64, 1 wave/row)
__global__ void k_l2norm(const float* __restrict__ in, float* __restrict__ out, float scale) {
    int row = blockIdx.x, lane = threadIdx.x;
    float a = in[row * 64 + lane] * scale;
    float b = in[row * 64 + lane + 32] * scale;
    float s = a * a + b * b;
    for (int m = 16; m; m >>= 1) s += __shfl_xor(s, m, 32);
    float nrm = fmaxf(sqrtf(s), 1e-12f);
    out[row * 64 + lane]      = a / nrm;
    out[row * 64 + lane + 32] = b / nrm;
}

__global__ void k_rowsq(const float* __restrict__ in, float* __restrict__ sq) {
    int row = blockIdx.x, lane = threadIdx.x;
    float a = in[row * 64 + lane], b = in[row * 64 + lane + 32];
    float s = a * a + b * b;
    for (int m = 16; m; m >>= 1) s += __shfl_xor(s, m, 32);
    if (lane == 0) sq[row] = s;
}

__global__ void k_align(const float* __restrict__ X, const float* __restrict__ Y,
                        float* __restrict__ acc) {
    int row = blockIdx.x, lane = threadIdx.x;
    float d0 = X[row * 64 + lane]      - Y[row * 64 + lane];
    float d1 = X[row * 64 + lane + 32] - Y[row * 64 + lane + 32];
    float s = d0 * d0 + d1 * d1;
    for (int m = 16; m; m >>= 1) s += __shfl_xor(s, m, 32);
    if (lane == 0) atomicAdd(acc, s);
}

// ---------------------------------------------------------------- WMMA fragment load
// A frag (16x32 f16, MxK) and B frag (32x16 = (X[N x K])^T) share the same per-lane
// pattern on CDNA5: lane L -> matrix row L%16; halves 0..7 at k = kb + (L/16)*8,
// halves 8..15 at k = kb + 16 + (L/16)*8.  Row-major source with stride K.
__device__ __forceinline__ v16h load_frag16(const _Float16* __restrict__ p,
                                            int stride, int row0, int col0) {
    int lane = threadIdx.x & 31;
    const _Float16* q = p + (long)(row0 + (lane & 15)) * stride + col0 + ((lane >> 4) << 3);
    v8h lo = *(const v8h*)q;
    v8h hi = *(const v8h*)(q + 16);
    v16h r;
#pragma unroll
    for (int i = 0; i < 8; ++i) { r[i] = lo[i]; r[i + 8] = hi[i]; }
    return r;
}

// ---------------------------------------------------------------- uniformity loss (Gram + sum-exp)
// grid (128,128), 1 wave/block; tile (ti,tj) of X X^T; upper-triangular only.
__global__ void k_lunif(const _Float16* __restrict__ Xh, const float* __restrict__ sq,
                        float* __restrict__ acc, float t) {
    int ti = blockIdx.x, tj = blockIdx.y;
    if (tj < ti) return;                     // tile fully below diagonal
    v8f c = {};
#pragma unroll
    for (int kb = 0; kb < EMBED; kb += 32) {
        v16h a = load_frag16(Xh, EMBED, ti * 16, kb);
        v16h b = load_frag16(Xh, EMBED, tj * 16, kb);
        c = __builtin_amdgcn_wmma_f32_16x16x32_f16(false, a, false, b, (short)0, c, false, false);
    }
    int lane = threadIdx.x & 31;
    int n = tj * 16 + (lane & 15);
    float sqn = sq[n];
    float s = 0.f;
#pragma unroll
    for (int v = 0; v < 8; ++v) {
        int m = ti * 16 + v + ((lane >> 4) << 3);
        if (n > m) {
            float d2 = fmaxf(sq[m] + sqn - 2.f * c[v], 0.f);
            s += expf(-t * d2);
        }
    }
    for (int msk = 16; msk; msk >>= 1) s += __shfl_xor(s, msk, 32);
    if (lane == 0) atomicAdd(acc, s);
}

// ---------------------------------------------------------------- NT GEMM: D = A[MxK] * B[NxK]^T + bias (opt relu)
// grid (M/16, N/16), 1 wave/block
__global__ void k_gemm_nt(const _Float16* __restrict__ A, const _Float16* __restrict__ B,
                          const float* __restrict__ bias, float* __restrict__ D,
                          int N, int K, int relu) {
    int ti = blockIdx.x, tj = blockIdx.y;
    v8f c = {};
    for (int kb = 0; kb < K; kb += 32) {
        v16h a = load_frag16(A, K, ti * 16, kb);
        v16h b = load_frag16(B, K, tj * 16, kb);
        c = __builtin_amdgcn_wmma_f32_16x16x32_f16(false, a, false, b, (short)0, c, false, false);
    }
    int lane = threadIdx.x & 31;
    int n = tj * 16 + (lane & 15);
    float bn = bias[n];
#pragma unroll
    for (int v = 0; v < 8; ++v) {
        int m = ti * 16 + v + ((lane >> 4) << 3);
        float x = c[v] + bn;
        if (relu) x = fmaxf(x, 0.f);
        D[(long)m * N + n] = x;
    }
}

// ---------------------------------------------------------------- finalize 3 outputs
__global__ void k_final(const float* __restrict__ S, float* __restrict__ out) {
    out[0] = S[0] / (float)BATCH;
    float cnt = (float)((long)BATCH * (BATCH - 1) / 2);
    out[1] = 0.5f * ((logf(S[1]) - logf(cnt)) + (logf(S[2]) - logf(cnt)));
    out[2] = S[3] / (float)BATCH;
}

// ================================================================ host
extern "C" void kernel_launch(void* const* d_in, const int* in_sizes, int n_in,
                              void* d_out, int out_size, void* d_ws, size_t ws_size,
                              hipStream_t stream) {
    (void)n_in; (void)out_size; (void)ws_size;
    const int*   users = (const int*)d_in[0];
    const int*   items = (const int*)d_in[1];
    const int*   erow  = (const int*)d_in[2];
    const int*   ecol  = (const int*)d_in[3];
    const float* evalv = (const float*)d_in[4];
    const float* uemb  = (const float*)d_in[5];
    const float* iemb  = (const float*)d_in[6];
    const float* nz1   = (const float*)d_in[7];
    const float* nz2   = (const float*)d_in[8];
    const float* W1    = (const float*)d_in[9];
    const float* b1    = (const float*)d_in[10];
    const float* W2    = (const float*)d_in[11];
    const float* b2    = (const float*)d_in[12];
    const int nE = in_sizes[2];

    const long NN64 = (long)N_NODES * EMBED;
    const long B64  = (long)BATCH * EMBED;
    const long BI   = (long)BATCH * INNER;

    float* fp  = (float*)d_ws;
    float* e    = fp; fp += NN64;
    float* enx  = fp; fp += NN64;
    float* accU = fp; fp += B64;
    float* accI = fp; fp += B64;
    float* ue   = fp; fp += B64;
    float* ie   = fp; fp += B64;
    float* u1   = fp; fp += B64;  float* u2 = fp; fp += B64;
    float* i1   = fp; fp += B64;  float* i2 = fp; fp += B64;
    float* un1  = fp; fp += B64;  float* un2 = fp; fp += B64;
    float* in1  = fp; fp += B64;  float* in2 = fp; fp += B64;
    float* pu1  = fp; fp += B64;  float* pu2 = fp; fp += B64;
    float* pi1  = fp; fp += B64;  float* pi2 = fp; fp += B64;
    float* hid  = fp; fp += BI;
    float* prd  = fp; fp += B64;
    float* sqb  = fp; fp += BATCH;
    float* S    = fp; fp += 64;
    _Float16* hp  = (_Float16*)fp;
    _Float16* Ah  = hp; hp += BI;            // x or hidden in f16
    _Float16* W1h = hp; hp += (long)INNER * EMBED;
    _Float16* W2h = hp; hp += (long)EMBED * INNER;
    _Float16* Ueh = hp; hp += B64;
    _Float16* Ieh = hp; hp += B64;

    auto NB = [](long n, int b) { return (unsigned)((n + b - 1) / b); };

    // scalars + weight conversion
    k_zero<<<1, 64, 0, stream>>>(S, 64);
    k_f32_to_f16<<<NB((long)INNER * EMBED, 256), 256, 0, stream>>>(W1, W1h, (long)INNER * EMBED);
    k_f32_to_f16<<<NB((long)EMBED * INNER, 256), 256, 0, stream>>>(W2, W2h, (long)EMBED * INNER);

    // ---------------- pass runner: noise == nullptr -> clean ----------------
    auto run_pass = [&](const float* noise) {
        k_zero<<<NB(B64, 256), 256, 0, stream>>>(accU, B64);
        k_zero<<<NB(B64, 256), 256, 0, stream>>>(accI, B64);
        k_concat<<<NB(NN64, 256), 256, 0, stream>>>(uemb, iemb, e);
        float* cur = e; float* nxt = enx;
        for (int l = 0; l < LAYERS; ++l) {
            k_zero<<<NB(NN64, 256), 256, 0, stream>>>(nxt, NN64);
            k_spmm<<<NB((long)nE * 32, 256), 256, 0, stream>>>(erow, ecol, evalv, cur, nxt, nE);
            if (noise)
                k_noise<<<NB(NN64, 256), 256, 0, stream>>>(nxt, noise, NN64);
            k_gather_add<<<NB(B64, 256), 256, 0, stream>>>(nxt, users, 0, accU);
            k_gather_add<<<NB(B64, 256), 256, 0, stream>>>(nxt, items, NUM_USERS, accI);
            float* t = cur; cur = nxt; nxt = t;
        }
    };

    // ---------------- clean pass: align + uniformity ----------------
    run_pass(nullptr);
    k_l2norm<<<BATCH, 32, 0, stream>>>(accU, ue, 1.f / LAYERS);
    k_l2norm<<<BATCH, 32, 0, stream>>>(accI, ie, 1.f / LAYERS);
    k_align<<<BATCH, 32, 0, stream>>>(ue, ie, &S[0]);

    k_f32_to_f16<<<NB(B64, 256), 256, 0, stream>>>(ue, Ueh, B64);
    k_rowsq<<<BATCH, 32, 0, stream>>>(ue, sqb);
    k_lunif<<<dim3(BATCH / 16, BATCH / 16), 32, 0, stream>>>(Ueh, sqb, &S[1], 2.0f);
    k_f32_to_f16<<<NB(B64, 256), 256, 0, stream>>>(ie, Ieh, B64);
    k_rowsq<<<BATCH, 32, 0, stream>>>(ie, sqb);
    k_lunif<<<dim3(BATCH / 16, BATCH / 16), 32, 0, stream>>>(Ieh, sqb, &S[2], 2.0f);

    // ---------------- two noisy passes ----------------
    run_pass(nz1);
    k_scale<<<NB(B64, 256), 256, 0, stream>>>(accU, u1, 1.f / LAYERS, B64);
    k_scale<<<NB(B64, 256), 256, 0, stream>>>(accI, i1, 1.f / LAYERS, B64);
    run_pass(nz2);
    k_scale<<<NB(B64, 256), 256, 0, stream>>>(accU, u2, 1.f / LAYERS, B64);
    k_scale<<<NB(B64, 256), 256, 0, stream>>>(accI, i2, 1.f / LAYERS, B64);

    // ---------------- predictor MLP (WMMA) + normalize ----------------
    auto predict = [&](const float* x, float* outn) {
        k_f32_to_f16<<<NB(B64, 256), 256, 0, stream>>>(x, Ah, B64);
        k_gemm_nt<<<dim3(BATCH / 16, INNER / 16), 32, 0, stream>>>(Ah, W1h, b1, hid, INNER, EMBED, 1);
        k_f32_to_f16<<<NB(BI, 256), 256, 0, stream>>>(hid, Ah, BI);
        k_gemm_nt<<<dim3(BATCH / 16, EMBED / 16), 32, 0, stream>>>(Ah, W2h, b2, prd, EMBED, INNER, 0);
        k_l2norm<<<BATCH, 32, 0, stream>>>(prd, outn, 1.f);
    };
    predict(u1, pu1); predict(u2, pu2); predict(i1, pi1); predict(i2, pi2);

    k_l2norm<<<BATCH, 32, 0, stream>>>(u1, un1, 1.f);
    k_l2norm<<<BATCH, 32, 0, stream>>>(u2, un2, 1.f);
    k_l2norm<<<BATCH, 32, 0, stream>>>(i1, in1, 1.f);
    k_l2norm<<<BATCH, 32, 0, stream>>>(i2, in2, 1.f);

    k_align<<<BATCH, 32, 0, stream>>>(un1, pu2, &S[3]);
    k_align<<<BATCH, 32, 0, stream>>>(un2, pu1, &S[3]);
    k_align<<<BATCH, 32, 0, stream>>>(in1, pi2, &S[3]);
    k_align<<<BATCH, 32, 0, stream>>>(in2, pi1, &S[3]);

    k_final<<<1, 1, 0, stream>>>(S, (float*)d_out);
}
